// InceptionDQ_51479478010582
// MI455X (gfx1250) — matrix operand order repaired and checked by reading, not compiled, not run on hardware
//
#include <hip/hip_runtime.h>

typedef _Float16 v8h  __attribute__((ext_vector_type(8)));
typedef _Float16 v16h __attribute__((ext_vector_type(16)));
typedef float    v8f  __attribute__((ext_vector_type(8)));

#define BN_EPS 1e-3f

// ---------------------------------------------------------------------------
// Pack NCHW f32 input -> NHWC f16 (K-contiguous for implicit GEMM)
// ---------------------------------------------------------------------------
__global__ void pack_x_kernel(const float* __restrict__ x,
                              _Float16* __restrict__ X0, int total) {
  int i = blockIdx.x * blockDim.x + threadIdx.x;
  if (i >= total) return;
  int w = i % 17; int t = i / 17;
  int h = t % 17; t /= 17;
  int c = t % 768; int b = t / 768;
  X0[((size_t)((b * 17 + h) * 17 + w)) * 768 + c] = (_Float16)x[i];
}

// ---------------------------------------------------------------------------
// Fake-quantize weights (symmetric per-tensor, 8-bit) and repack OIHW f32 ->
// [K/64][OCpad][64] f16, K = (kh*KW + kw)*IC + ic. OC zero-padded to OCpad.
// ---------------------------------------------------------------------------
__global__ void pack_w_kernel(const float* __restrict__ w,
                              const float* __restrict__ sw,
                              _Float16* __restrict__ Wp,
                              int OC, int ICc, int KH, int KW, int OCpad,
                              int total) {
  int i = blockIdx.x * blockDim.x + threadIdx.x;
  if (i >= total) return;
  int kin = i & 63;
  int n   = (i >> 6) % OCpad;
  int kb  = i / (64 * OCpad);
  int k   = kb * 64 + kin;
  _Float16 val = (_Float16)0.f;
  if (n < OC) {
    int r  = k / (KW * ICc);
    int s  = (k / ICc) % KW;
    int ic = k % ICc;
    float scale = sw[0];
    float wn = w[((n * ICc + ic) * KH + r) * KW + s] / scale;
    wn = fminf(fmaxf(wn, -1.f), 1.f);
    val = (_Float16)(rintf(wn * 127.f) * (1.f / 127.f) * scale);
  }
  Wp[i] = val;
}

// ---------------------------------------------------------------------------
// Implicit-GEMM conv + BN + ReLU + PACT act-quant, f16 WMMA.
// All geometry compile-time: no integer-division chains, K-step 64
// (8 x v_wmma_f32_16x16x32_f16 per barrier pair).
//   M = 128*OH*OW pixels (multiple of 64), N = OCPAD, K = KH*KW*IC (mult 64).
//   8 wave32s; tile 64(M) x 128(N); each wave 32x32 via 4 f32 accumulators.
// MODE 0: store NHWC f16 to Yh.  MODE 1: store NCHW f32 into d_out at CHOFF.
// ---------------------------------------------------------------------------
template <int IC, int KH, int KW, int OH, int OW, int STRD, int PADH, int PADW,
          int OC, int OCPAD, int MODE, int CHOFF>
__global__ __launch_bounds__(256)
void qconv_wmma_kernel(const _Float16* __restrict__ Xin,
                       const _Float16* __restrict__ Wp,
                       const float* __restrict__ gamma,
                       const float* __restrict__ beta,
                       const float* __restrict__ mean,
                       const float* __restrict__ var,
                       const float* __restrict__ alpha,
                       _Float16* __restrict__ Yh, float* __restrict__ Yf) {
  constexpr int IH = 17, IW = 17;
  constexpr int KDIM = IC * KH * KW;       // multiple of 64
  constexpr int OHOW = OH * OW;
  constexpr int LAS = 72;                  // LDS row stride (64 + 8 pad) halfs

  __shared__ __align__(16) _Float16 lA[64 * LAS];   // 64 pixels x 64 K
  __shared__ __align__(16) _Float16 lB[128 * LAS];  // 128 n x 64 K

  const int tid  = threadIdx.x;
  const int lane = tid & 31;
  const int wid  = tid >> 5;
  const int wm   = wid >> 2;   // 0..1 : M sub-tile
  const int wn   = wid & 3;    // 0..3 : N sub-tile
  const int tileM = blockIdx.y * 64;
  const int tileN = blockIdx.x * 128;

  // A-stage geometry: thread loads 16 halfs (two b128) of one pixel row.
  const int arow = tid >> 2;        // 0..63
  const int acol = (tid & 3) * 16;  // 0,16,32,48
  const int m    = tileM + arow;
  const int bb   = m / OHOW;        // constant divisor -> mul/shift
  const int rm   = m % OHOW;
  const int ohb  = (rm / OW) * STRD - PADH;
  const int owb  = (rm % OW) * STRD - PADW;

  // B-stage geometry: thread loads 32 halfs (four b128) of one weight row.
  const int brow = tid >> 1;            // local n 0..127
  const int bofs = (tid & 1) * 32;

  v8f acc00 = {0.f, 0.f, 0.f, 0.f, 0.f, 0.f, 0.f, 0.f};
  v8f acc01 = acc00, acc10 = acc00, acc11 = acc00;

  for (int k0 = 0; k0 < KDIM; k0 += 64) {
    // ---- stage A tile (64 x 64 f16) from NHWC input, zero-fill padding ----
    int r, s, ic0;
    if constexpr (KH == 1 && KW == 1) {
      r = 0; s = 0; ic0 = k0;
    } else {
      r   = k0 / (KW * IC);        // compile-time divisors
      s   = (k0 / IC) % KW;
      ic0 = k0 % IC;
    }
    const int ih = ohb + r, iw = owb + s;
    v8h av0 = {(_Float16)0.f, (_Float16)0.f, (_Float16)0.f, (_Float16)0.f,
               (_Float16)0.f, (_Float16)0.f, (_Float16)0.f, (_Float16)0.f};
    v8h av1 = av0;
    if (ih >= 0 && ih < IH && iw >= 0 && iw < IW) {
      const _Float16* xp =
          Xin + ((size_t)((bb * IH + ih) * IW + iw)) * IC + ic0 + acol;
      av0 = *(const v8h*)(xp);
      av1 = *(const v8h*)(xp + 8);
    }
    *(v8h*)(lA + arow * LAS + acol)     = av0;
    *(v8h*)(lA + arow * LAS + acol + 8) = av1;

    // ---- stage B tile (128 n x 64 K f16) from packed [K/64][OCPAD][64] ----
    const _Float16* wp =
        Wp + ((size_t)(k0 >> 6) * OCPAD + (tileN + brow)) * 64 + bofs;
    *(v8h*)(lB + brow * LAS + bofs)      = *(const v8h*)(wp);
    *(v8h*)(lB + brow * LAS + bofs + 8)  = *(const v8h*)(wp + 8);
    *(v8h*)(lB + brow * LAS + bofs + 16) = *(const v8h*)(wp + 16);
    *(v8h*)(lB + brow * LAS + bofs + 24) = *(const v8h*)(wp + 24);
    __builtin_prefetch(wp + (size_t)OCPAD * 64, 0, 1);  // next K-block
    __syncthreads();

    // ---- 2 K-substeps x 4 WMMA, fragments per ISA 7.12.2 lane layouts ----
    const int mrow = lane & 15;
    const int koff = (lane >> 4) * 8;   // A: elems 0..7=K[koff..], 8..15=K[koff+16..]
    const int bko  = (lane >> 4) * 8;   // B: elems 0..7=K[bko..], 8..15=K[bko+16..]
    const _Float16* pa0 = lA + (wm * 32 + mrow) * LAS;
    const _Float16* pa1 = pa0 + 16 * LAS;
    const _Float16* pb0 = lB + (wn * 32 + mrow) * LAS;
    const _Float16* pb1 = pb0 + 16 * LAS;
#pragma unroll
    for (int ks = 0; ks < 64; ks += 32) {
      v16h fa0 = __builtin_shufflevector(
          *(const v8h*)(pa0 + ks + koff), *(const v8h*)(pa0 + ks + 16 + koff),
          0, 1, 2, 3, 4, 5, 6, 7, 8, 9, 10, 11, 12, 13, 14, 15);
      v16h fa1 = __builtin_shufflevector(
          *(const v8h*)(pa1 + ks + koff), *(const v8h*)(pa1 + ks + 16 + koff),
          0, 1, 2, 3, 4, 5, 6, 7, 8, 9, 10, 11, 12, 13, 14, 15);
      v16h fb0 = __builtin_shufflevector(
          *(const v8h*)(pb0 + ks + bko), *(const v8h*)(pb0 + ks + bko + 16),
          0, 1, 2, 3, 4, 5, 6, 7, 8, 9, 10, 11, 12, 13, 14, 15);
      v16h fb1 = __builtin_shufflevector(
          *(const v8h*)(pb1 + ks + bko), *(const v8h*)(pb1 + ks + bko + 16),
          0, 1, 2, 3, 4, 5, 6, 7, 8, 9, 10, 11, 12, 13, 14, 15);
      acc00 = __builtin_amdgcn_wmma_f32_16x16x32_f16(false, fa0, false, fb0,
                                                     (short)0, acc00, false, false);
      acc01 = __builtin_amdgcn_wmma_f32_16x16x32_f16(false, fa0, false, fb1,
                                                     (short)0, acc01, false, false);
      acc10 = __builtin_amdgcn_wmma_f32_16x16x32_f16(false, fa1, false, fb0,
                                                     (short)0, acc10, false, false);
      acc11 = __builtin_amdgcn_wmma_f32_16x16x32_f16(false, fa1, false, fb1,
                                                     (short)0, acc11, false, false);
    }
    __syncthreads();
  }

  // ---- epilogue: BN(running stats) + ReLU + PACT fake-quant ----
  const float al = alpha[0];
  const float qs = 255.f / al;
  const float dq = al * (1.f / 255.f);
  const int mhi = (lane >> 4) * 8;  // C/D layout: lanes 16-31 hold M+8

#define QEPI(ACC, MS, NS)                                                    \
  do {                                                                       \
    int n = tileN + wn * 32 + (NS) * 16 + (lane & 15);                       \
    if (n < OC) {                                                            \
      float sc = gamma[n] * rsqrtf(var[n] + BN_EPS);                         \
      float sh = beta[n] - mean[n] * sc;                                     \
      _Pragma("unroll") for (int i = 0; i < 8; ++i) {                        \
        int gm = tileM + wm * 32 + (MS) * 16 + mhi + i;                      \
        float y = ACC[i] * sc + sh;                                          \
        y = fminf(fmaxf(y, 0.f), al);                                        \
        y = rintf(y * qs) * dq;                                              \
        if constexpr (MODE == 0) {                                           \
          Yh[(size_t)gm * OC + n] = (_Float16)y;                             \
        } else {                                                             \
          int bo = gm / OHOW; int sp = gm % OHOW;                            \
          Yf[((size_t)bo * 1280 + CHOFF + n) * OHOW + sp] = y;               \
        }                                                                    \
      }                                                                      \
    }                                                                        \
  } while (0)

  QEPI(acc00, 0, 0);
  QEPI(acc01, 0, 1);
  QEPI(acc10, 1, 0);
  QEPI(acc11, 1, 1);
#undef QEPI
}

// ---------------------------------------------------------------------------
// 3x3 stride-2 valid maxpool, NCHW f32 -> concat channels [512,1280)
// ---------------------------------------------------------------------------
__global__ void maxpool_kernel(const float* __restrict__ x,
                               float* __restrict__ out, int total) {
  int i = blockIdx.x * blockDim.x + threadIdx.x;
  if (i >= total) return;
  int ow = i & 7;
  int oh = (i >> 3) & 7;
  int c  = (i >> 6) % 768;
  int b  = i / (64 * 768);
  const float* p = x + ((size_t)(b * 768 + c) * 17 + oh * 2) * 17 + ow * 2;
  float mx = -3.402823466e+38f;
  for (int r = 0; r < 3; ++r)
    for (int s = 0; s < 3; ++s) mx = fmaxf(mx, p[r * 17 + s]);
  out[((size_t)b * 1280 + 512 + c) * 64 + oh * 8 + ow] = mx;
}

__global__ void alpha_kernel(const float* __restrict__ a1,
                             const float* __restrict__ a2,
                             const float* __restrict__ xa,
                             float* __restrict__ out, int pos) {
  out[pos] = fmaxf(fmaxf(a1[0], a2[0]), xa[0]);
}

// ---------------------------------------------------------------------------
// Host launcher
// ---------------------------------------------------------------------------
extern "C" void kernel_launch(void* const* d_in, const int* in_sizes, int n_in,
                              void* d_out, int out_size, void* d_ws,
                              size_t ws_size, hipStream_t stream) {
  const float* x       = (const float*)d_in[0];
  const float* x_alpha = (const float*)d_in[1];
  // params flatten: 6 layers x {w, sw, gamma, beta, mean, var, alpha}
  auto P = [&](int layer, int field) -> const float* {
    return (const float*)d_in[2 + layer * 7 + field];
  };

  _Float16* ws = (_Float16*)d_ws;
  // workspace layout (units: halfs)
  const size_t oX0 = 0;                                    // 36992 x 768
  const size_t oY1 = oX0 + (size_t)36992 * 768;            // 36992 x 192
  const size_t oC1 = oY1 + (size_t)36992 * 192;
  const size_t oC2 = oC1 + (size_t)36992 * 192;
  const size_t oC3 = oC2 + (size_t)36992 * 192;
  const size_t oW1 = oC3 + (size_t)36992 * 192;            // 256 x 768
  const size_t oW2 = oW1 + (size_t)256 * 768;              // 384 x 1728
  const size_t oW3 = oW2 + (size_t)384 * 1728;             // 256 x 768
  const size_t oW4 = oW3 + (size_t)256 * 768;              // 256 x 1344
  const size_t oW5 = oW4 + (size_t)256 * 1344;             // 256 x 1344
  const size_t oW6 = oW5 + (size_t)256 * 1344;             // 256 x 1728

  // 1) input NCHW f32 -> NHWC f16
  {
    int total = 128 * 768 * 17 * 17;
    pack_x_kernel<<<(total + 255) / 256, 256, 0, stream>>>(x, ws + oX0, total);
  }

  // 2) fake-quant + repack weights ([K/64][OCpad][64])
  struct WDesc { int layer, OC, IC, KH, KW, OCpad; size_t off; };
  const WDesc wd[6] = {
      {0, 192, 768, 1, 1, 256, oW1}, {1, 320, 192, 3, 3, 384, oW2},
      {2, 192, 768, 1, 1, 256, oW3}, {3, 192, 192, 1, 7, 256, oW4},
      {4, 192, 192, 7, 1, 256, oW5}, {5, 192, 192, 3, 3, 256, oW6}};
  for (int i = 0; i < 6; ++i) {
    int Kdim  = wd[i].IC * wd[i].KH * wd[i].KW;
    int total = (Kdim / 64) * wd[i].OCpad * 64;
    pack_w_kernel<<<(total + 255) / 256, 256, 0, stream>>>(
        P(wd[i].layer, 0), P(wd[i].layer, 1), ws + wd[i].off, wd[i].OC,
        wd[i].IC, wd[i].KH, wd[i].KW, wd[i].OCpad, total);
  }

  float* out = (float*)d_out;

#define LAUNCH_CONV(IC_, KH_, KW_, OH_, OW_, ST_, PH_, PW_, OC_, OCP_, MODE_, \
                    CHOFF_, LAYER_, XIN_, WOFF_, YH_)                         \
  do {                                                                        \
    dim3 grid((OCP_) / 128, (128 * (OH_) * (OW_)) / 64);                      \
    qconv_wmma_kernel<IC_, KH_, KW_, OH_, OW_, ST_, PH_, PW_, OC_, OCP_,      \
                      MODE_, CHOFF_><<<grid, 256, 0, stream>>>(               \
        (XIN_), ws + (WOFF_), P(LAYER_, 2), P(LAYER_, 3), P(LAYER_, 4),       \
        P(LAYER_, 5), P(LAYER_, 6), (YH_), out);                              \
  } while (0)

  // branch3x3: 1x1 -> 3x3/s2 (concat ch 0..319)
  LAUNCH_CONV(768, 1, 1, 17, 17, 1, 0, 0, 192, 256, 0, 0, 0, ws + oX0, oW1,
              ws + oY1);
  LAUNCH_CONV(192, 3, 3, 8, 8, 2, 0, 0, 320, 384, 1, 0, 1, ws + oY1, oW2,
              (_Float16*)nullptr);
  // branch7x7x3: 1x1 -> 1x7 -> 7x1 -> 3x3/s2 (concat ch 320..511)
  LAUNCH_CONV(768, 1, 1, 17, 17, 1, 0, 0, 192, 256, 0, 0, 2, ws + oX0, oW3,
              ws + oC1);
  LAUNCH_CONV(192, 1, 7, 17, 17, 1, 0, 3, 192, 256, 0, 0, 3, ws + oC1, oW4,
              ws + oC2);
  LAUNCH_CONV(192, 7, 1, 17, 17, 1, 3, 0, 192, 256, 0, 0, 4, ws + oC2, oW5,
              ws + oC3);
  LAUNCH_CONV(192, 3, 3, 8, 8, 2, 0, 0, 192, 256, 1, 320, 5, ws + oC3, oW6,
              (_Float16*)nullptr);
#undef LAUNCH_CONV

  // pool branch (concat ch 512..1279)
  {
    int total = 128 * 768 * 8 * 8;
    maxpool_kernel<<<(total + 255) / 256, 256, 0, stream>>>(x, out, total);
  }

  // output alpha = max(alpha_b3x3_2, alpha_b7_4, x_alpha), detached
  alpha_kernel<<<1, 1, 0, stream>>>(P(1, 6), P(5, 6), x_alpha, out,
                                    out_size - 1);
}
